// BertClassifierv4_74191265071198
// MI455X (gfx1250) — compile-verified, matrix operand order checked by reading, not run on hardware
//
#include <hip/hip_runtime.h>

// ---------------------------------------------------------------------------
// BERT-style NLI classifier head for gfx1250 (MI455X).
// Heavy GEMMs (QKV projections, QK^T scores, hs@hs^T similarity) run on
// v_wmma_f32_16x16x32_bf16. Softmax-consumer GEMMs are algebraically
// collapsed to probability column-sums (only masked means are needed).
// Next-K-step fragments are warmed with global_prefetch_b8 hints.
// ---------------------------------------------------------------------------

#define Bb 32
#define Ss 512
#define Hh 768
#define NH 8
#define HD 96
#define NEGV (-1.0e9f)

typedef __bf16 bf16_t;
typedef bf16_t v16bf __attribute__((ext_vector_type(16)));
typedef bf16_t v8bf  __attribute__((ext_vector_type(8)));
typedef float  v8f   __attribute__((ext_vector_type(8)));

__device__ __forceinline__ float telu_f(float x) { return x * tanhf(__expf(x)); }

// A-fragment loader per CDNA5 ISA 16-bit 16x32 layout:
// lane<16 : row=lane,    K in [k0..k0+8) U [k0+16..k0+24)
// lane>=16: row=lane-16, K in [k0+8..k0+16) U [k0+24..k0+32)
__device__ __forceinline__ v16bf load_frag_bf16(const bf16_t* __restrict__ base,
                                                int ld, int k0, int lane) {
  const int row = lane & 15;
  const int kb  = k0 + ((lane & 16) ? 8 : 0);
  const bf16_t* p = base + row * ld + kb;
  v8bf lo = *(const v8bf*)(p);
  v8bf hi = *(const v8bf*)(p + 16);
  v16bf r;
#pragma unroll
  for (int i = 0; i < 8; ++i) { r[i] = lo[i]; r[i + 8] = hi[i]; }
  return r;
}

// Per-lane prefetch of the cachelines a future fragment load will touch.
// Lowers to global_prefetch_b8 (no LOADcnt activity).
__device__ __forceinline__ void prefetch_frag(const bf16_t* __restrict__ base,
                                              int ld, int k0, int lane) {
  const int row = lane & 15;
  const int kb  = k0 + ((lane & 16) ? 8 : 0);
  __builtin_prefetch((const void*)(base + row * ld + kb), 0, 1);
}

__device__ __forceinline__ v8f wmma_bf16(v16bf a, v16bf b, v8f c) {
  return __builtin_amdgcn_wmma_f32_16x16x32_bf16(false, a, false, b, (short)0, c,
                                                 false, false);
}

// ---------------------------------------------------------------------------
// Mask / segment extraction
// ---------------------------------------------------------------------------
__global__ void mask_kernel(const int* __restrict__ ids, const int* __restrict__ am,
                            float* __restrict__ prem, float* __restrict__ hyp,
                            float* __restrict__ cnts) {
  const int b = blockIdx.x, s = threadIdx.x;  // 512 threads
  __shared__ int s1s, s2s, amc;
  if (s == 0) { s1s = Ss; s2s = -1; amc = 0; }
  __syncthreads();
  const int id = ids[b * Ss + s];
  if (id == 102) { atomicMin(&s1s, s); atomicMax(&s2s, s); }
  if (am[b * Ss + s] != 0) atomicAdd(&amc, 1);
  __syncthreads();
  const int s1 = s1s, s2 = s2s;
  prem[b * Ss + s] = (s >= 1 && s < s1) ? 1.f : 0.f;
  hyp[b * Ss + s]  = (s > s1 && s < s2) ? 1.f : 0.f;
  if (s == 0) {
    cnts[b * 4 + 0] = fmaxf((float)amc, 1e-9f);
    cnts[b * 4 + 1] = fmaxf((float)(s1 - 1), 1e-9f);
    cnts[b * 4 + 2] = fmaxf((float)(s2 - s1 - 1), 1e-9f);
  }
}

__global__ void f32_to_bf16_kernel(const float* __restrict__ in,
                                   bf16_t* __restrict__ out, int n) {
  const int i = blockIdx.x * blockDim.x + threadIdx.x;
  if (i < n) out[i] = (bf16_t)in[i];
}

// Wt[n*R + k] = W[k*C + n]  (R = K-dim rows of W, C = N-dim cols)
__global__ void transpose_bf16_kernel(const float* __restrict__ W,
                                      bf16_t* __restrict__ Wt, int R, int C) {
  const int i = blockIdx.x * blockDim.x + threadIdx.x;
  if (i < R * C) {
    const int c = i / R, r = i % R;
    Wt[i] = (bf16_t)W[r * C + c];
  }
}

// ---------------------------------------------------------------------------
// WMMA GEMM:  out[M,N] = A[M,K] * Bt[N,K]^T + bias[n]   (bf16 in, bf16 out)
// block = 128 threads (4 waves); block tile 64x64; wave tile 16x64.
// ---------------------------------------------------------------------------
__global__ void gemm_proj_kernel(const bf16_t* __restrict__ A,
                                 const bf16_t* __restrict__ Bt,
                                 const float* __restrict__ bias,
                                 bf16_t* __restrict__ out,
                                 int M, int N, int Kd) {
  const int lane = threadIdx.x & 31;
  const int wave = threadIdx.x >> 5;
  const int m0 = blockIdx.y * 64 + wave * 16;
  const int n0 = blockIdx.x * 64;
  v8f c[4] = {};
  for (int kk = 0; kk < Kd; kk += 32) {
    // warm next K-step while this one computes
    if (kk + 32 < Kd) {
      prefetch_frag(A + (size_t)m0 * Kd, Kd, kk + 32, lane);
#pragma unroll
      for (int j = 0; j < 4; ++j)
        prefetch_frag(Bt + (size_t)(n0 + j * 16) * Kd, Kd, kk + 32, lane);
    }
    v16bf a = load_frag_bf16(A + (size_t)m0 * Kd, Kd, kk, lane);
#pragma unroll
    for (int j = 0; j < 4; ++j) {
      v16bf bfr = load_frag_bf16(Bt + (size_t)(n0 + j * 16) * Kd, Kd, kk, lane);
      c[j] = wmma_bf16(a, bfr, c[j]);
    }
  }
  const int nc = lane & 15;
#pragma unroll
  for (int j = 0; j < 4; ++j) {
#pragma unroll
    for (int r = 0; r < 8; ++r) {
      const int m = m0 + ((lane < 16) ? r : r + 8);
      const int n = n0 + j * 16 + nc;
      out[(size_t)m * N + n] = (bf16_t)(c[j][r] + bias[n]);
    }
  }
}

// ---------------------------------------------------------------------------
// Attention: per (b,h,q-tile): scores = Q Kt^T / sqrt(HD), mask keys by prem,
// softmax over keys, accumulate hyp[s]-weighted probability column sums.
// ---------------------------------------------------------------------------
__global__ void attn_colsum_kernel(const bf16_t* __restrict__ Q,
                                   const bf16_t* __restrict__ K,
                                   const float* __restrict__ prem,
                                   const float* __restrict__ hyp,
                                   float* __restrict__ asum) {
  const int b = blockIdx.z, h = blockIdx.y, q0 = blockIdx.x * 16;
  const int tid = threadIdx.x, lane = tid & 31, wave = tid >> 5;
  __shared__ float sc[16][Ss];
  __shared__ float rmx[16], rinv[16], rw[16];

  const bf16_t* Qb = Q + (size_t)(b * Ss + q0) * Hh + h * HD;
  const bf16_t* Kb = K + (size_t)b * Ss * Hh + h * HD;

  v8f c[8] = {};
  for (int kk = 0; kk < HD; kk += 32) {
    v16bf a = load_frag_bf16(Qb, Hh, kk, lane);
#pragma unroll
    for (int j = 0; j < 8; ++j) {
      const int nt = wave * 8 + j;
      if (j + 1 < 8 && kk == 0)
        prefetch_frag(Kb + (size_t)(nt + 1) * 16 * Hh, Hh, 0, lane);
      v16bf bfr = load_frag_bf16(Kb + (size_t)nt * 16 * Hh, Hh, kk, lane);
      c[j] = wmma_bf16(a, bfr, c[j]);
    }
  }
  const float scale = 0.1020620726f;  // 1/sqrt(96)
  const int nc = lane & 15;
#pragma unroll
  for (int j = 0; j < 8; ++j) {
    const int t0 = (wave * 8 + j) * 16 + nc;
#pragma unroll
    for (int r = 0; r < 8; ++r) {
      const int m = (lane < 16) ? r : r + 8;
      sc[m][t0] = c[j][r] * scale;
    }
  }
  __syncthreads();

  // softmax stats: 16 rows x 8 threads each
  const int r = tid >> 3, sub = tid & 7;
  const float* pm = prem + b * Ss;
  float mx = -3.4e38f;
  for (int t = sub; t < Ss; t += 8) {
    const float v = (pm[t] > 0.f) ? sc[r][t] : NEGV;
    mx = fmaxf(mx, v);
  }
  mx = fmaxf(mx, __shfl_xor(mx, 1));
  mx = fmaxf(mx, __shfl_xor(mx, 2));
  mx = fmaxf(mx, __shfl_xor(mx, 4));
  float sm = 0.f;
  for (int t = sub; t < Ss; t += 8) {
    const float v = (pm[t] > 0.f) ? sc[r][t] : NEGV;
    sm += __expf(v - mx);
  }
  sm += __shfl_xor(sm, 1);
  sm += __shfl_xor(sm, 2);
  sm += __shfl_xor(sm, 4);
  if (sub == 0) { rmx[r] = mx; rinv[r] = 1.f / sm; rw[r] = hyp[b * Ss + q0 + r]; }
  __syncthreads();

  float* as = asum + (size_t)(b * NH + h) * Ss;
  for (int t = tid; t < Ss; t += 128) {
    const bool pk = pm[t] > 0.f;
    float acc = 0.f;
#pragma unroll
    for (int rr = 0; rr < 16; ++rr) {
      const float v = pk ? sc[rr][t] : NEGV;
      acc += rw[rr] * __expf(v - rmx[rr]) * rinv[rr];
    }
    atomicAdd(&as[t], acc);
  }
}

// ---------------------------------------------------------------------------
// Similarity: per (b,q-tile): sim = hs hs^T (K=768), two masked softmaxes
// from one strip: (mask=hyp, weight=prem) -> cs1 ; (mask=prem, weight=hyp) -> cs2
// ---------------------------------------------------------------------------
__global__ void sim_colsum_kernel(const bf16_t* __restrict__ HS,
                                  const float* __restrict__ prem,
                                  const float* __restrict__ hyp,
                                  float* __restrict__ cs1,
                                  float* __restrict__ cs2) {
  const int b = blockIdx.y, q0 = blockIdx.x * 16;
  const int tid = threadIdx.x, lane = tid & 31, wave = tid >> 5;
  __shared__ float sc[16][Ss];
  __shared__ float rmx1[16], rinv1[16], rw1[16];
  __shared__ float rmx2[16], rinv2[16], rw2[16];

  const bf16_t* Qb = HS + (size_t)(b * Ss + q0) * Hh;
  const bf16_t* Kb = HS + (size_t)b * Ss * Hh;

  v8f c[8] = {};
  for (int kk = 0; kk < Hh; kk += 32) {
    if (kk + 32 < Hh) prefetch_frag(Qb, Hh, kk + 32, lane);
    v16bf a = load_frag_bf16(Qb, Hh, kk, lane);
#pragma unroll
    for (int j = 0; j < 8; ++j) {
      const int nt = wave * 8 + j;
      if (kk + 32 < Hh)
        prefetch_frag(Kb + (size_t)nt * 16 * Hh, Hh, kk + 32, lane);
      v16bf bfr = load_frag_bf16(Kb + (size_t)nt * 16 * Hh, Hh, kk, lane);
      c[j] = wmma_bf16(a, bfr, c[j]);
    }
  }
  const int nc = lane & 15;
#pragma unroll
  for (int j = 0; j < 8; ++j) {
    const int t0 = (wave * 8 + j) * 16 + nc;
#pragma unroll
    for (int r = 0; r < 8; ++r) {
      const int m = (lane < 16) ? r : r + 8;
      sc[m][t0] = c[j][r];
    }
  }
  __syncthreads();

  const int r = tid >> 3, sub = tid & 7;
  const float* pm = prem + b * Ss;
  const float* hm = hyp + b * Ss;
  float mxa = -3.4e38f, mxb = -3.4e38f;
  for (int t = sub; t < Ss; t += 8) {
    const float v = sc[r][t];
    mxa = fmaxf(mxa, (hm[t] > 0.f) ? v : NEGV);
    mxb = fmaxf(mxb, (pm[t] > 0.f) ? v : NEGV);
  }
  mxa = fmaxf(mxa, __shfl_xor(mxa, 1)); mxa = fmaxf(mxa, __shfl_xor(mxa, 2)); mxa = fmaxf(mxa, __shfl_xor(mxa, 4));
  mxb = fmaxf(mxb, __shfl_xor(mxb, 1)); mxb = fmaxf(mxb, __shfl_xor(mxb, 2)); mxb = fmaxf(mxb, __shfl_xor(mxb, 4));
  float sa = 0.f, sb = 0.f;
  for (int t = sub; t < Ss; t += 8) {
    const float v = sc[r][t];
    sa += __expf(((hm[t] > 0.f) ? v : NEGV) - mxa);
    sb += __expf(((pm[t] > 0.f) ? v : NEGV) - mxb);
  }
  sa += __shfl_xor(sa, 1); sa += __shfl_xor(sa, 2); sa += __shfl_xor(sa, 4);
  sb += __shfl_xor(sb, 1); sb += __shfl_xor(sb, 2); sb += __shfl_xor(sb, 4);
  if (sub == 0) {
    rmx1[r] = mxa; rinv1[r] = 1.f / sa; rw1[r] = pm[q0 + r];
    rmx2[r] = mxb; rinv2[r] = 1.f / sb; rw2[r] = hm[q0 + r];
  }
  __syncthreads();

  for (int t = tid; t < Ss; t += 128) {
    const bool m1 = hm[t] > 0.f, m2 = pm[t] > 0.f;
    float a1 = 0.f, a2 = 0.f;
#pragma unroll
    for (int rr = 0; rr < 16; ++rr) {
      const float v = sc[rr][t];
      a1 += rw1[rr] * __expf((m1 ? v : NEGV) - rmx1[rr]) * rinv1[rr];
      a2 += rw2[rr] * __expf((m2 ? v : NEGV) - rmx2[rr]) * rinv2[rr];
    }
    atomicAdd(&cs1[b * Ss + t], a1);
    atomicAdd(&cs2[b * Ss + t], a2);
  }
}

// ---------------------------------------------------------------------------
// Pooled features: x0 = hs[:,0] + masked-mean(am) + masked-max(am);
// sdiff = |mmean(prem) - mmean(hyp)|
// ---------------------------------------------------------------------------
__global__ void pool_kernel(const float* __restrict__ hs, const int* __restrict__ am,
                            const float* __restrict__ prem, const float* __restrict__ hyp,
                            const float* __restrict__ cnts,
                            float* __restrict__ x0, float* __restrict__ sdiff) {
  const int b = blockIdx.y;
  const int hh = blockIdx.x * 128 + threadIdx.x;
  const float* base = hs + (size_t)b * Ss * Hh + hh;
  float sm = 0.f, mx = NEGV, sp = 0.f, sh = 0.f;
  for (int s = 0; s < Ss; ++s) {
    const float x = base[(size_t)s * Hh];
    if (am[b * Ss + s] != 0) { sm += x; mx = fmaxf(mx, x); }
    sp += prem[b * Ss + s] * x;
    sh += hyp[b * Ss + s] * x;
  }
  const float amc = cnts[b * 4 + 0], pc = cnts[b * 4 + 1], hc = cnts[b * 4 + 2];
  x0[b * Hh + hh] = base[0] + sm / amc + mx;
  sdiff[b * Hh + hh] = fabsf(sp / pc - sh / hc);
}

// ---------------------------------------------------------------------------
// Feature MLP: h = LN(x0 @ fe_w1 + b1) * g + be ; feat = telu(h @ fe_w2 + b2)
// ---------------------------------------------------------------------------
__global__ void feat_kernel(const float* __restrict__ x0,
                            const float* __restrict__ w1, const float* __restrict__ b1,
                            const float* __restrict__ g, const float* __restrict__ be,
                            const float* __restrict__ w2, const float* __restrict__ b2,
                            float* __restrict__ combined) {
  const int b = blockIdx.x, tid = threadIdx.x;  // 512 threads
  __shared__ float xs[Hh], h1[512], red[512];
  __shared__ float mean_s, var_s;
  for (int i = tid; i < Hh; i += 512) xs[i] = x0[b * Hh + i];
  __syncthreads();
  float acc = b1[tid];
  for (int i = 0; i < Hh; ++i) acc += xs[i] * w1[i * 512 + tid];
  red[tid] = acc;
  __syncthreads();
  for (int st = 256; st > 0; st >>= 1) { if (tid < st) red[tid] += red[tid + st]; __syncthreads(); }
  if (tid == 0) mean_s = red[0] / 512.f;
  __syncthreads();
  red[tid] = acc * acc;
  __syncthreads();
  for (int st = 256; st > 0; st >>= 1) { if (tid < st) red[tid] += red[tid + st]; __syncthreads(); }
  if (tid == 0) var_s = red[0] / 512.f - mean_s * mean_s;
  __syncthreads();
  h1[tid] = (acc - mean_s) * rsqrtf(var_s + 1e-5f) * g[tid] + be[tid];
  __syncthreads();
  if (tid < 128) {
    float f = b2[tid];
    for (int j = 0; j < 512; ++j) f += h1[j] * w2[j * 128 + tid];
    combined[b * 512 + tid] = telu_f(f);
  }
}

// out[b, j] = telu(in[b,:] @ W[:,j] + bias[j])  (128 outputs)
__global__ void proj128_telu_kernel(const float* __restrict__ in,
                                    const float* __restrict__ W,
                                    const float* __restrict__ bias,
                                    float* __restrict__ out,
                                    int D, int instride, int outstride) {
  const int b = blockIdx.x, j = threadIdx.x;  // 128 threads
  const float* x = in + (size_t)b * instride;
  float acc = bias[j];
  for (int i = 0; i < D; ++i) acc += x[i] * W[i * 128 + j];
  out[(size_t)b * outstride + j] = telu_f(acc);
}

// al1[d] = cs1 . hs[:,d] / prem_cnt ; al2[d] = cs2 . hs[:,d] / hyp_cnt
__global__ void al_kernel(const float* __restrict__ hs,
                          const float* __restrict__ cs1, const float* __restrict__ cs2,
                          const float* __restrict__ cnts, float* __restrict__ alcat) {
  const int b = blockIdx.x, d = threadIdx.x;  // 768 threads
  float a1 = 0.f, a2 = 0.f;
  for (int t = 0; t < Ss; ++t) {
    const float x = hs[((size_t)b * Ss + t) * Hh + d];
    a1 += cs1[b * Ss + t] * x;
    a2 += cs2[b * Ss + t] * x;
  }
  alcat[b * 1536 + d]       = a1 / cnts[b * 4 + 1];
  alcat[b * 1536 + Hh + d]  = a2 / cnts[b * 4 + 2];
}

__global__ void align_kernel(const float* __restrict__ alcat,
                             const float* __restrict__ w1, const float* __restrict__ b1,
                             const float* __restrict__ w2, const float* __restrict__ b2,
                             float* __restrict__ combined) {
  const int b = blockIdx.x, tid = threadIdx.x;  // 768 threads
  __shared__ float xs[1536], t1[Hh];
  for (int i = tid; i < 1536; i += Hh) xs[i] = alcat[b * 1536 + i];
  __syncthreads();
  float acc = b1[tid];
  for (int i = 0; i < 1536; ++i) acc += xs[i] * w1[i * Hh + tid];
  t1[tid] = acc;
  __syncthreads();
  if (tid < 128) {
    float f = b2[tid];
    for (int j = 0; j < Hh; ++j) f += t1[j] * w2[j * 128 + tid];
    combined[b * 512 + 384 + tid] = telu_f(f);
  }
}

// vbar[n] = (asum[b,h,:] . V[:,n]) / hyp_cnt ; ctxm = vbar @ wo + bo ;
// attn_rep = telu(ctxm @ ap_w + ap_b)
__global__ void attnrep_kernel(const bf16_t* __restrict__ V,
                               const float* __restrict__ asum,
                               const float* __restrict__ cnts,
                               const float* __restrict__ wo, const float* __restrict__ bo,
                               const float* __restrict__ apw, const float* __restrict__ apb,
                               float* __restrict__ combined) {
  const int b = blockIdx.x, tid = threadIdx.x;  // 768 threads
  __shared__ float vbar[Hh], ctxm[Hh];
  const int h = tid / HD;
  const float hc = cnts[b * 4 + 2];
  const float* as = asum + (size_t)(b * NH + h) * Ss;
  const bf16_t* Vb = V + (size_t)b * Ss * Hh + tid;
  float acc = 0.f;
  for (int t = 0; t < Ss; ++t) acc += as[t] * (float)Vb[(size_t)t * Hh];
  vbar[tid] = acc / hc;
  __syncthreads();
  float c2 = bo[tid];
  for (int i = 0; i < Hh; ++i) c2 += vbar[i] * wo[i * Hh + tid];
  ctxm[tid] = c2;
  __syncthreads();
  if (tid < 128) {
    float f = apb[tid];
    for (int i = 0; i < Hh; ++i) f += ctxm[i] * apw[i * 128 + tid];
    combined[b * 512 + 256 + tid] = telu_f(f);
  }
}

__global__ void cls_kernel(const float* __restrict__ combined,
                           const float* __restrict__ w1, const float* __restrict__ b1,
                           const float* __restrict__ w2, const float* __restrict__ b2,
                           float* __restrict__ out) {
  const int b = blockIdx.x, tid = threadIdx.x;  // 64 threads
  __shared__ float xs[512], c1[64];
  for (int i = tid; i < 512; i += 64) xs[i] = combined[b * 512 + i];
  __syncthreads();
  float acc = b1[tid];
  for (int i = 0; i < 512; ++i) acc += xs[i] * w1[i * 64 + tid];
  c1[tid] = telu_f(acc);
  __syncthreads();
  if (tid < 3) {
    float l = b2[tid];
    for (int j = 0; j < 64; ++j) l += c1[j] * w2[j * 3 + tid];
    out[b * 3 + tid] = l;
  }
}

// ---------------------------------------------------------------------------
extern "C" void kernel_launch(void* const* d_in, const int* in_sizes, int n_in,
                              void* d_out, int out_size, void* d_ws, size_t ws_size,
                              hipStream_t stream) {
  const float* hs   = (const float*)d_in[0];
  const int*   ids  = (const int*)d_in[1];
  const int*   amsk = (const int*)d_in[2];
  const float* fe_w1 = (const float*)d_in[3];
  const float* fe_b1 = (const float*)d_in[4];
  const float* fe_g  = (const float*)d_in[5];
  const float* fe_be = (const float*)d_in[6];
  const float* fe_w2 = (const float*)d_in[7];
  const float* fe_b2 = (const float*)d_in[8];
  const float* al_w1 = (const float*)d_in[9];
  const float* al_b1 = (const float*)d_in[10];
  const float* al_w2 = (const float*)d_in[11];
  const float* al_b2 = (const float*)d_in[12];
  const float* wq = (const float*)d_in[13];
  const float* bq = (const float*)d_in[14];
  const float* wk = (const float*)d_in[15];
  const float* bk = (const float*)d_in[16];
  const float* wv = (const float*)d_in[17];
  const float* bv = (const float*)d_in[18];
  const float* wo = (const float*)d_in[19];
  const float* bo = (const float*)d_in[20];
  const float* dp_w = (const float*)d_in[21];
  const float* dp_b = (const float*)d_in[22];
  const float* ap_w = (const float*)d_in[23];
  const float* ap_b = (const float*)d_in[24];
  const float* cl_w1 = (const float*)d_in[25];
  const float* cl_b1 = (const float*)d_in[26];
  const float* cl_w2 = (const float*)d_in[27];
  const float* cl_b2 = (const float*)d_in[28];

  char* ws = (char*)d_ws;
  size_t off = 0;
  auto alloc = [&](size_t bytes) -> char* {
    char* p = ws + off;
    off = (off + bytes + 255) & ~(size_t)255;
    return p;
  };
  const size_t nHS = (size_t)Bb * Ss * Hh;          // 12,582,912
  bf16_t* hs_bf = (bf16_t*)alloc(nHS * 2);
  bf16_t* wqt   = (bf16_t*)alloc((size_t)Hh * Hh * 2);
  bf16_t* wkt   = (bf16_t*)alloc((size_t)Hh * Hh * 2);
  bf16_t* wvt   = (bf16_t*)alloc((size_t)Hh * Hh * 2);
  bf16_t* Qp    = (bf16_t*)alloc(nHS * 2);
  bf16_t* Kp    = (bf16_t*)alloc(nHS * 2);
  bf16_t* Vp    = (bf16_t*)alloc(nHS * 2);
  float* prem   = (float*)alloc((size_t)Bb * Ss * 4);
  float* hyp    = (float*)alloc((size_t)Bb * Ss * 4);
  float* cnts   = (float*)alloc((size_t)Bb * 4 * 4);
  float* x0     = (float*)alloc((size_t)Bb * Hh * 4);
  float* sdiff  = (float*)alloc((size_t)Bb * Hh * 4);
  float* asum   = (float*)alloc((size_t)Bb * NH * Ss * 4);
  float* cs1    = (float*)alloc((size_t)Bb * Ss * 4);
  float* cs2    = (float*)alloc((size_t)Bb * Ss * 4);
  float* alcat  = (float*)alloc((size_t)Bb * 1536 * 4);
  float* comb   = (float*)alloc((size_t)Bb * 512 * 4);
  (void)ws_size; (void)in_sizes; (void)n_in; (void)out_size;

  // zero accumulators (graph-capture-safe)
  hipMemsetAsync(asum, 0, (size_t)Bb * NH * Ss * 4, stream);
  hipMemsetAsync(cs1, 0, (size_t)Bb * Ss * 4, stream);
  hipMemsetAsync(cs2, 0, (size_t)Bb * Ss * 4, stream);

  mask_kernel<<<Bb, Ss, 0, stream>>>(ids, amsk, prem, hyp, cnts);

  {
    const int n = (int)nHS;
    f32_to_bf16_kernel<<<(n + 255) / 256, 256, 0, stream>>>(hs, hs_bf, n);
  }
  {
    const int n = Hh * Hh;
    transpose_bf16_kernel<<<(n + 255) / 256, 256, 0, stream>>>(wq, wqt, Hh, Hh);
    transpose_bf16_kernel<<<(n + 255) / 256, 256, 0, stream>>>(wk, wkt, Hh, Hh);
    transpose_bf16_kernel<<<(n + 255) / 256, 256, 0, stream>>>(wv, wvt, Hh, Hh);
  }

  // QKV projections: M=16384, N=768, K=768 on WMMA
  {
    dim3 grid(Hh / 64, (Bb * Ss) / 64);
    gemm_proj_kernel<<<grid, 128, 0, stream>>>(hs_bf, wqt, bq, Qp, Bb * Ss, Hh, Hh);
    gemm_proj_kernel<<<grid, 128, 0, stream>>>(hs_bf, wkt, bk, Kp, Bb * Ss, Hh, Hh);
    gemm_proj_kernel<<<grid, 128, 0, stream>>>(hs_bf, wvt, bv, Vp, Bb * Ss, Hh, Hh);
  }

  pool_kernel<<<dim3(Hh / 128, Bb), 128, 0, stream>>>(hs, amsk, prem, hyp, cnts, x0, sdiff);

  attn_colsum_kernel<<<dim3(Ss / 16, NH, Bb), 128, 0, stream>>>(Qp, Kp, prem, hyp, asum);
  sim_colsum_kernel<<<dim3(Ss / 16, Bb), 128, 0, stream>>>(hs_bf, prem, hyp, cs1, cs2);

  feat_kernel<<<Bb, 512, 0, stream>>>(x0, fe_w1, fe_b1, fe_g, fe_be, fe_w2, fe_b2, comb);
  proj128_telu_kernel<<<Bb, 128, 0, stream>>>(sdiff, dp_w, dp_b, comb + 128, Hh, Hh, 512);
  al_kernel<<<Bb, Hh, 0, stream>>>(hs, cs1, cs2, cnts, alcat);
  align_kernel<<<Bb, Hh, 0, stream>>>(alcat, al_w1, al_b1, al_w2, al_b2, comb);
  attnrep_kernel<<<Bb, Hh, 0, stream>>>(Vp, asum, cnts, wo, bo, ap_w, ap_b, comb);

  cls_kernel<<<Bb, 64, 0, stream>>>(comb, cl_w1, cl_b1, cl_w2, cl_b2, (float*)d_out);
}
